// GAT_86681029968549
// MI455X (gfx1250) — matrix operand order repaired and checked by reading, not compiled, not run on hardware
//
#include <hip/hip_runtime.h>
#include <hip/hip_bf16.h>

typedef __bf16 bf16;
typedef __attribute__((ext_vector_type(16))) __bf16 v16bf;
typedef __attribute__((ext_vector_type(8)))  __bf16 v8bf;
typedef __attribute__((ext_vector_type(8)))  float  v8f;

#define NEG_SLOPE 0.2f

// ---------------------------------------------------------------- utilities

__global__ void fill_f32(float* p, float v, long long n) {
    long long t = (long long)blockIdx.x * blockDim.x + threadIdx.x;
    if (t < n) p[t] = v;
}

__global__ void f32_to_bf16(const float* __restrict__ in, bf16* __restrict__ out, long long n) {
    long long t = (long long)blockIdx.x * blockDim.x + threadIdx.x;
    if (t < n) out[t] = (bf16)in[t];
}

__device__ inline void atomicMaxF(float* addr, float v) {
    if (v >= 0.f) atomicMax((int*)addr, __float_as_int(v));
    else          atomicMin((unsigned int*)addr, __float_as_uint(v));
}

__device__ inline void decode_edge(const int* __restrict__ ei, int E, int eid, int& s, int& d) {
    if (eid < E) { s = ei[eid]; d = ei[E + eid]; }
    else         { s = d = eid - E; }           // self-loops appended after edges
}

// ---------------------------------------------------------------- WMMA GEMM
// C[M,NC] (f32, row-major) = A[M,K] (bf16, row-major) * W[NC,K]^T (bf16, row-major)
// One wave computes a 16x64 output tile: A-frag reused across 4 B-tiles.
// A-frag layout (16x32 bf16): lane l -> row (l&15); halfs[0:8] = K  k0+hi*16..,
//   halfs[8:16] = K k0+hi*16+8.. (hi = l>>4).
// B-frag layout (32x16 bf16): lane l -> column (l&15); halfs[0:16] = 16
//   consecutive K starting at k0 + hi*16  => a single 32B load from W's row.
__global__ __launch_bounds__(256)
void gemm_bf16_wmma(const bf16* __restrict__ A, const bf16* __restrict__ W,
                    float* __restrict__ C, int M, int K, int NC, int totalWaves) {
    int gtid = blockIdx.x * blockDim.x + threadIdx.x;
    int wave = gtid >> 5;
    if (wave >= totalWaves) return;
    int lane = gtid & 31;
    int tilesN = NC >> 6;                       // 64-column groups
    int rowT = wave / tilesN;
    int colG = wave - rowT * tilesN;
    size_t m0 = (size_t)rowT * 16;
    int    n0 = colG * 64;
    int r  = lane & 15;
    int hi = lane >> 4;

    v8f acc0 = {}, acc1 = {}, acc2 = {}, acc3 = {};

    const bf16* arow  = A + (m0 + (size_t)r) * (size_t)K;
    const bf16* wrow0 = W + ((size_t)(n0 +  0 + r)) * (size_t)K + hi * 16;
    const bf16* wrow1 = W + ((size_t)(n0 + 16 + r)) * (size_t)K + hi * 16;
    const bf16* wrow2 = W + ((size_t)(n0 + 32 + r)) * (size_t)K + hi * 16;
    const bf16* wrow3 = W + ((size_t)(n0 + 48 + r)) * (size_t)K + hi * 16;

    for (int k0 = 0; k0 < K; k0 += 32) {
        v8bf alo = *(const v8bf*)(arow + k0 + hi * 16);
        v8bf ahi = *(const v8bf*)(arow + k0 + hi * 16 + 8);
        v16bf a;
#pragma unroll
        for (int i = 0; i < 8; ++i) { a[i] = alo[i]; a[i + 8] = ahi[i]; }

        v16bf b0 = *(const v16bf*)(wrow0 + k0);
        v16bf b1 = *(const v16bf*)(wrow1 + k0);
        v16bf b2 = *(const v16bf*)(wrow2 + k0);
        v16bf b3 = *(const v16bf*)(wrow3 + k0);

        acc0 = __builtin_amdgcn_wmma_f32_16x16x32_bf16(false, a, false, b0, (short)0, acc0, false, false);
        acc1 = __builtin_amdgcn_wmma_f32_16x16x32_bf16(false, a, false, b1, (short)0, acc1, false, false);
        acc2 = __builtin_amdgcn_wmma_f32_16x16x32_bf16(false, a, false, b2, (short)0, acc2, false, false);
        acc3 = __builtin_amdgcn_wmma_f32_16x16x32_bf16(false, a, false, b3, (short)0, acc3, false, false);
    }

    // C/D layout: lane l, vgpr v -> (M = v + hi*8, N = l&15)
    float* crow = C + (m0 + (size_t)hi * 8) * (size_t)NC + n0 + r;
#pragma unroll
    for (int v = 0; v < 8; ++v) {
        crow[(size_t)v * NC +  0] = acc0[v];
        crow[(size_t)v * NC + 16] = acc1[v];
        crow[(size_t)v * NC + 32] = acc2[v];
        crow[(size_t)v * NC + 48] = acc3[v];
    }
}

// ---------------------------------------------------------------- attention

// alpha_src[n,h] = sum_c h[n,h,c]*att_src[h,c]; same for dst.
__global__ void node_alpha(const float* __restrict__ h, const float* __restrict__ ws,
                           const float* __restrict__ wd, float* __restrict__ as,
                           float* __restrict__ ad, int nN, int H, int Cc) {
    int t = blockIdx.x * blockDim.x + threadIdx.x;
    if (t >= nN * H) return;
    int n = t / H, hh = t - n * H;
    const float* hp = h + ((size_t)n * H + hh) * Cc;
    const float* pw = ws + (size_t)hh * Cc;
    const float* pd = wd + (size_t)hh * Cc;
    float ss = 0.f, sd = 0.f;
    for (int c = 0; c < Cc; ++c) { float v = hp[c]; ss += v * pw[c]; sd += v * pd[c]; }
    as[t] = ss; ad[t] = sd;
}

__device__ inline float edge_e(const float* as, const float* ad, int s, int d, int H, int hh) {
    float v = as[(size_t)s * H + hh] + ad[(size_t)d * H + hh];
    return v > 0.f ? v : v * NEG_SLOPE;
}

__global__ void edge_max_k(const int* __restrict__ ei, int E, int Etot,
                           const float* __restrict__ as, const float* __restrict__ ad,
                           float* __restrict__ emax, int H) {
    int t = blockIdx.x * blockDim.x + threadIdx.x;
    if (t >= Etot * H) return;
    int e = t / H, hh = t - e * H;
    int s, d; decode_edge(ei, E, e, s, d);
    atomicMaxF(&emax[(size_t)d * H + hh], edge_e(as, ad, s, d, H, hh));
}

__global__ void edge_expsum_k(const int* __restrict__ ei, int E, int Etot,
                              const float* __restrict__ as, const float* __restrict__ ad,
                              const float* __restrict__ emax, float* __restrict__ exbuf,
                              float* __restrict__ denom, int H) {
    int t = blockIdx.x * blockDim.x + threadIdx.x;
    if (t >= Etot * H) return;
    int e = t / H, hh = t - e * H;
    int s, d; decode_edge(ei, E, e, s, d);
    float v  = edge_e(as, ad, s, d, H, hh);
    float ex = __expf(v - emax[(size_t)d * H + hh]);
    exbuf[t] = ex;
    atomicAdd(&denom[(size_t)d * H + hh], ex);
}

__global__ void edge_norm_k(const int* __restrict__ ei, int E, int Etot,
                            const float* __restrict__ denom, float* __restrict__ exbuf, int H) {
    int t = blockIdx.x * blockDim.x + threadIdx.x;
    if (t >= Etot * H) return;
    int e = t / H, hh = t - e * H;
    int s, d; decode_edge(ei, E, e, s, d);
    exbuf[t] = exbuf[t] / (denom[(size_t)d * H + hh] + 1e-16f);
}

// out[dst, h, c] += h[src, h, c] * alpha[e, h]
__global__ void edge_scatter_k(const int* __restrict__ ei, int E, long long Etot,
                               const float* __restrict__ h, const float* __restrict__ alpha,
                               float* __restrict__ out, int H, int Cc) {
    long long t = (long long)blockIdx.x * blockDim.x + threadIdx.x;
    int HC = H * Cc;
    if (t >= Etot * HC) return;
    int e   = (int)(t / HC);
    int idx = (int)(t - (long long)e * HC);
    int hh  = idx / Cc;
    int s, d; decode_edge(ei, E, e, s, d);
    float w = alpha[(size_t)e * H + hh];
    atomicAdd(&out[(size_t)d * HC + idx], h[(size_t)s * HC + idx] * w);
}

// ---------------------------------------------------------------- epilogues

__global__ void bias_elu_to_bf16(const float* __restrict__ in, const float* __restrict__ b,
                                 bf16* __restrict__ outb, long long n, int stride) {
    long long t = (long long)blockIdx.x * blockDim.x + threadIdx.x;
    if (t >= n) return;
    float v = in[t] + b[(int)(t % stride)];
    v = v > 0.f ? v : (__expf(v) - 1.f);
    outb[t] = (bf16)v;
}

__global__ void bias_add_inplace(float* __restrict__ out, const float* __restrict__ b,
                                 long long n, int stride) {
    long long t = (long long)blockIdx.x * blockDim.x + threadIdx.x;
    if (t >= n) return;
    out[t] += b[(int)(t % stride)];
}

// ---------------------------------------------------------------- launcher

extern "C" void kernel_launch(void* const* d_in, const int* in_sizes, int n_in,
                              void* d_out, int out_size, void* d_ws, size_t ws_size,
                              hipStream_t stream) {
    const float* x    = (const float*)d_in[0];
    const int*   ei   = (const int*)  d_in[1];
    const float* W1   = (const float*)d_in[2];
    const float* asr1 = (const float*)d_in[3];
    const float* adt1 = (const float*)d_in[4];
    const float* b1   = (const float*)d_in[5];
    const float* W2   = (const float*)d_in[6];
    const float* asr2 = (const float*)d_in[7];
    const float* adt2 = (const float*)d_in[8];
    const float* b2   = (const float*)d_in[9];
    float* out = (float*)d_out;

    const int IN_C = 256, H1 = 8, C1 = 64, HC1 = 512, C2 = 64;
    const long long N    = in_sizes[0] / IN_C;     // 50000
    const long long E    = in_sizes[1] / 2;        // 500000
    const long long Etot = E + N;                  // 550000

    // -------- workspace carve-up (256B aligned), with aliasing ----------
    char* ws = (char*)d_ws;
    size_t off = 0;
    auto carve = [&](size_t bytes) { size_t r = off; off = (off + bytes + 255) & ~(size_t)255; return r; };
    size_t o_xb    = carve((size_t)N * IN_C * 2);      // x in bf16    (dead after GEMM1)
    size_t o_w1b   = carve((size_t)HC1 * IN_C * 2);
    size_t o_w2b   = carve((size_t)C2 * HC1 * 2);
    size_t o_h1    = carve((size_t)N * HC1 * 4);       // layer1 linear out (f32)
    size_t o_out1  = carve((size_t)N * HC1 * 4);       // layer1 aggregation
    size_t o_h1b   = carve((size_t)N * HC1 * 2);       // elu(out1+b1) in bf16
    size_t o_as1   = carve((size_t)N * H1 * 4);
    size_t o_ad1   = carve((size_t)N * H1 * 4);
    size_t o_emax1 = carve((size_t)N * H1 * 4);
    size_t o_den1  = carve((size_t)N * H1 * 4);
    // aliases (lifetimes disjoint):
    size_t o_ex1   = o_xb;                              // Etot*H1*4 = 17.6MB <= 25.6MB
    size_t o_h2    = o_h1;                              // layer2 lives after layer1 scatter
    size_t o_as2   = o_h2   + (((size_t)N * C2 * 4 + 255) & ~(size_t)255);
    size_t o_ad2   = o_as2  + (((size_t)N * 4 + 255) & ~(size_t)255);
    size_t o_emax2 = o_ad2  + (((size_t)N * 4 + 255) & ~(size_t)255);
    size_t o_den2  = o_emax2+ (((size_t)N * 4 + 255) & ~(size_t)255);
    size_t o_ex2   = o_den2 + (((size_t)N * 4 + 255) & ~(size_t)255);

    bf16*  xb   = (bf16*)(ws + o_xb);
    bf16*  w1b  = (bf16*)(ws + o_w1b);
    bf16*  w2b  = (bf16*)(ws + o_w2b);
    float* h1   = (float*)(ws + o_h1);
    float* out1 = (float*)(ws + o_out1);
    bf16*  h1b  = (bf16*)(ws + o_h1b);
    float* as1  = (float*)(ws + o_as1);
    float* ad1  = (float*)(ws + o_ad1);
    float* em1  = (float*)(ws + o_emax1);
    float* dn1  = (float*)(ws + o_den1);
    float* ex1  = (float*)(ws + o_ex1);
    float* h2   = (float*)(ws + o_h2);
    float* as2  = (float*)(ws + o_as2);
    float* ad2  = (float*)(ws + o_ad2);
    float* em2  = (float*)(ws + o_emax2);
    float* dn2  = (float*)(ws + o_den2);
    float* ex2  = (float*)(ws + o_ex2);

    const int B = 256;
    auto blocks = [&](long long n) { return (unsigned)((n + B - 1) / B); };
    const float NEG_INF = -__builtin_huge_valf();

    // -------- conversions --------
    f32_to_bf16<<<blocks(N * IN_C), B, 0, stream>>>(x,  xb,  N * IN_C);
    f32_to_bf16<<<blocks(HC1 * IN_C), B, 0, stream>>>(W1, w1b, (long long)HC1 * IN_C);
    f32_to_bf16<<<blocks(C2 * HC1), B, 0, stream>>>(W2, w2b, (long long)C2 * HC1);

    // ======================= layer 1 =======================
    fill_f32<<<blocks(N * H1), B, 0, stream>>>(em1, NEG_INF, N * H1);
    fill_f32<<<blocks(N * H1), B, 0, stream>>>(dn1, 0.f,     N * H1);
    fill_f32<<<blocks(N * HC1), B, 0, stream>>>(out1, 0.f,   N * HC1);

    int waves1 = (int)(N / 16) * (HC1 / 64);
    gemm_bf16_wmma<<<blocks((long long)waves1 * 32), B, 0, stream>>>(
        xb, w1b, h1, (int)N, IN_C, HC1, waves1);

    node_alpha<<<blocks(N * H1), B, 0, stream>>>(h1, asr1, adt1, as1, ad1, (int)N, H1, C1);
    edge_max_k   <<<blocks(Etot * H1), B, 0, stream>>>((const int*)ei, (int)E, (int)Etot, as1, ad1, em1, H1);
    edge_expsum_k<<<blocks(Etot * H1), B, 0, stream>>>((const int*)ei, (int)E, (int)Etot, as1, ad1, em1, ex1, dn1, H1);
    edge_norm_k  <<<blocks(Etot * H1), B, 0, stream>>>((const int*)ei, (int)E, (int)Etot, dn1, ex1, H1);
    edge_scatter_k<<<blocks(Etot * HC1), B, 0, stream>>>((const int*)ei, (int)E, Etot, h1, ex1, out1, H1, C1);

    bias_elu_to_bf16<<<blocks(N * HC1), B, 0, stream>>>(out1, b1, h1b, N * HC1, HC1);

    // ======================= layer 2 =======================
    fill_f32<<<blocks(N), B, 0, stream>>>(em2, NEG_INF, N);
    fill_f32<<<blocks(N), B, 0, stream>>>(dn2, 0.f,     N);
    fill_f32<<<blocks(N * C2), B, 0, stream>>>(out, 0.f, N * C2);

    int waves2 = (int)(N / 16) * (C2 / 64);
    gemm_bf16_wmma<<<blocks((long long)waves2 * 32), B, 0, stream>>>(
        h1b, w2b, h2, (int)N, HC1, C2, waves2);

    node_alpha<<<blocks(N * 1), B, 0, stream>>>(h2, asr2, adt2, as2, ad2, (int)N, 1, C2);
    edge_max_k   <<<blocks(Etot), B, 0, stream>>>((const int*)ei, (int)E, (int)Etot, as2, ad2, em2, 1);
    edge_expsum_k<<<blocks(Etot), B, 0, stream>>>((const int*)ei, (int)E, (int)Etot, as2, ad2, em2, ex2, dn2, 1);
    edge_norm_k  <<<blocks(Etot), B, 0, stream>>>((const int*)ei, (int)E, (int)Etot, dn2, ex2, 1);
    edge_scatter_k<<<blocks(Etot * C2), B, 0, stream>>>((const int*)ei, (int)E, Etot, h2, ex2, out, 1, C2);

    bias_add_inplace<<<blocks(N * C2), B, 0, stream>>>(out, b2, N * C2, C2);
    (void)n_in; (void)out_size; (void)ws_size;
}
